// CTCPrefixSearch_7756710937361
// MI455X (gfx1250) — compile-verified, matrix operand order checked by reading, not run on hardware
//
#include <hip/hip_runtime.h>
#include <stdint.h>

// ---------------- problem constants (from reference) ----------------
#define N_    128
#define KP_   32
#define V_    2048
#define S_    200
#define W_    32
#define CAND_ (KP_ * V_)        // 65536 extension candidates per row
#define NC_   (CAND_ + KP_)     // + 32 non-extension candidates = 65568

// d_out layout (flat f32, reference tuple order)
#define O_YNEXT   0                                  // (S+1, N, W) = 823296
#define O_LAST    ((S_ + 1) * N_ * W_)               // 823296
#define O_LENS    (O_LAST + N_ * W_)                 // 827392
#define O_NB      (O_LENS + N_ * W_)                 // 831488
#define O_B       (O_NB + N_ * W_)                   // 835584
#define O_ISPFX   (O_B + N_ * W_)                    // 839680  (N, W, W) = 131072
#define O_SRC     (O_ISPFX + N_ * W_ * W_)           // 970752
#define O_NONEXT  (O_SRC + N_ * W_)                  // 974848

typedef __attribute__((ext_vector_type(16))) _Float16 v16h;
typedef __attribute__((ext_vector_type(8)))  float    v8f;

// =====================================================================
// Kernel A: one wave (32 lanes) per row n.
//   - per-beam tables, has-match bitmask (scatter of ext_is_exact),
//   - b_nonext / nb_nonext terms,
//   - absorb column-sum via V_WMMA_F32_16X16X32_F16: colsum = ones(16x32) x M(32x32)
// =====================================================================
__global__ __launch_bounds__(32) void ctc_advA(
    const float* __restrict__ ext, const float* __restrict__ nonext,
    const float* __restrict__ blank, const float* __restrict__ nbp,
    const float* __restrict__ bp, const int* __restrict__ yprev,
    const int* __restrict__ ylast, const int* __restrict__ ylens,
    const int* __restrict__ pfx,
    uint32_t* __restrict__ ws_mask, float* __restrict__ ws_nbnon,
    float* __restrict__ ws_bnon)
{
    const int n = blockIdx.x;
    const int lane = threadIdx.x;            // wave32: 0..31

    __shared__ int   s_lens[KP_];
    __shared__ int   s_last[KP_];
    __shared__ float s_nb[KP_], s_b[KP_];
    __shared__ uint32_t s_mask[KP_ * 64];    // V_ bits per beam

    // per-beam scalars
    const int   L  = ylens[n * KP_ + lane];
    const int   yl = min(max(ylast[n * KP_ + lane], 0), V_ - 1);
    const float nb = nbp[n * KP_ + lane];
    const float b  = bp[n * KP_ + lane];
    s_lens[lane] = L; s_last[lane] = yl; s_nb[lane] = nb; s_b[lane] = b;
    __syncthreads();

    // has_match bitmask: lane k owns row k; to_match[n,k,j] = y_prev[clip(L_k), n, j]
    #pragma unroll 4
    for (int i = 0; i < 64; ++i) s_mask[lane * 64 + i] = 0u;
    {
        const int lclip = min(max(L, 0), S_ - 1);
        const int* yrow = yprev + (size_t)lclip * N_ * KP_ + n * KP_;
        const int* prow = pfx + ((size_t)n * KP_ + lane) * KP_;
        for (int j = 0; j < KP_; ++j) {
            const bool exact = ((L + 1) == s_lens[j]) && (prow[j] != 0);
            if (exact) {
                const int tok = min(max(yrow[j], 0), V_ - 1);
                s_mask[lane * 64 + (tok >> 5)] |= (1u << (tok & 31));
            }
        }
    }
    __syncthreads();
    {
        uint32_t* mrow = ws_mask + ((size_t)n * KP_ + lane) * 64;
        #pragma unroll 4
        for (int i = 0; i < 64; ++i) mrow[i] = s_mask[lane * 64 + i];
    }

    // ----- absorb colsums via WMMA -----
    // M[k][j] = ext_is_exact(k,j) ? ((tok==last_k ? 0 : nb_k) + b_k) * ext[n,k,tok] : 0
    // colsum[j] = sum_k M[k][j]  ==  (ones 16x32 f16) x (M 32x16 f16 half)
    // B as packed here (32x16, contiguous K): lanes 0..15 hold K=0..15, lanes 16..31 hold K=16..31,
    //                              column N = (lane & 15) [+ colbase], 2 K per VGPR.
    v16h bf[2];
    const int colpos = lane & 15;
    const int kbase  = (lane < 16) ? 0 : 16;
    #pragma unroll
    for (int h = 0; h < 2; ++h) {
        const int col = colpos + 16 * h;   // target beam j
        const int Lj  = s_lens[col];
        #pragma unroll
        for (int i = 0; i < 16; ++i) {
            const int k = kbase + i;       // source beam
            const int Lk = s_lens[k];
            const bool exact = ((Lk + 1) == Lj) &&
                               (pfx[((size_t)n * KP_ + k) * KP_ + col] != 0);
            float mval = 0.f;
            if (exact) {
                const int lclip = min(max(Lk, 0), S_ - 1);
                const int tok = min(max(yprev[(size_t)lclip * N_ * KP_ + n * KP_ + col], 0), V_ - 1);
                const float base = ((tok == s_last[k]) ? 0.f : s_nb[k]) + s_b[k];
                mval = base * ext[((size_t)n * KP_ + k) * V_ + tok];
            }
            bf[h][i] = (_Float16)mval;
        }
    }
    v16h av;
    #pragma unroll
    for (int i = 0; i < 16; ++i) av[i] = (_Float16)1.0f;
    v8f c = {};
    // D[m][j] = sum_k 1 * M[k][j]  (every row m carries the colsum)
    v8f d0 = __builtin_amdgcn_wmma_f32_16x16x32_f16(false, av, false, bf[0],
                                                    (short)0, c, false, false);
    v8f d1 = __builtin_amdgcn_wmma_f32_16x16x32_f16(false, av, false, bf[1],
                                                    (short)0, c, false, false);

    // b_nonext = (nb+b)*blank ; nb_nonext = nb*nonext[last] + colsum
    const float bl = blank[n];
    ws_bnon[n * KP_ + lane] = (nb + b) * bl;
    if (lane < 16) {
        const int j0 = lane, j1 = lane + 16;
        ws_nbnon[n * KP_ + j0] =
            s_nb[j0] * nonext[(size_t)n * V_ + s_last[j0]] + d0[0];
        ws_nbnon[n * KP_ + j1] =
            s_nb[j1] * nonext[(size_t)n * V_ + s_last[j1]] + d1[0];
    }
}

// =====================================================================
// Kernel B: one 512-thread block per row n.
//   async-DMA stage ext row (256KB) -> LDS, masked transform to candidate
//   scores in place, repeated block argmax x32 (wave-shuffle reduce),
//   emit scalar outputs + selection metadata for kernels C/D.
// =====================================================================
__global__ __launch_bounds__(512) void ctc_advB(
    const float* __restrict__ ext, const float* __restrict__ nbp,
    const float* __restrict__ bp, const int* __restrict__ ylast,
    const int* __restrict__ ylens, const uint32_t* __restrict__ ws_mask,
    const float* __restrict__ ws_nbnon, const float* __restrict__ ws_bnon,
    float* __restrict__ out,
    int* __restrict__ wsrc, int* __restrict__ wext, int* __restrict__ wplen,
    int* __restrict__ wlens, int* __restrict__ wnone)
{
    extern __shared__ unsigned char smem[];          // dynamic LDS, base offset 0
    float*    cand   = (float*)smem;                 // NC_ floats
    uint32_t* m      = (uint32_t*)(cand + NC_);      // KP_*64
    float*    s_base = (float*)(m + KP_ * 64);       // nb+b
    float*    s_b    = s_base + KP_;
    int*      s_last = (int*)(s_b + KP_);
    float*    s_nbn  = (float*)(s_last + KP_);
    float*    s_bn   = s_nbn + KP_;
    float*    wbest  = s_bn + KP_;                   // 16 wave maxima
    int*      wbidx  = (int*)(wbest + 16);
    float*    selV   = (float*)(wbidx + 16);         // W_ selections
    int*      selI   = (int*)(selV + W_);

    const int n = blockIdx.x, tid = threadIdx.x;

    // ---- async stage: 256KB of ext_probs_t row n -> LDS [0, CAND_*4) ----
    {
        const uint64_t sbase = (uint64_t)(const void*)(ext + (size_t)n * KP_ * V_);
        for (int i = tid * 4; i < CAND_; i += blockDim.x * 4) {
            uint32_t lds_off = (uint32_t)(i * 4);    // LDS byte address
            uint32_t g_off   = (uint32_t)(i * 4);    // global byte offset
            asm volatile("global_load_async_to_lds_b128 %0, %1, %2"
                         :: "v"(lds_off), "v"(g_off), "s"(sbase) : "memory");
        }
    }
    // small tables while DMA is in flight (disjoint LDS region)
    if (tid < KP_) {
        const float nb = nbp[n * KP_ + tid], b = bp[n * KP_ + tid];
        s_base[tid] = nb + b;
        s_b[tid]    = b;
        s_last[tid] = min(max(ylast[n * KP_ + tid], 0), V_ - 1);
        s_nbn[tid]  = ws_nbnon[n * KP_ + tid];
        s_bn[tid]   = ws_bnon[n * KP_ + tid];
    }
    for (int i = tid; i < KP_ * 64; i += blockDim.x)
        m[i] = ws_mask[(size_t)n * KP_ * 64 + i];

    asm volatile("s_wait_asynccnt 0x0" ::: "memory");
    __syncthreads();

    // ---- transform in place: cand = has_match ? -inf : base * ext ----
    for (int i = tid; i < CAND_; i += blockDim.x) {
        const int k = i >> 11, v = i & (V_ - 1);
        const float e  = cand[i];
        const bool  hm = (m[k * 64 + (v >> 5)] >> (v & 31)) & 1u;
        const float base = (v == s_last[k]) ? s_b[k] : s_base[k];
        cand[i] = hm ? -INFINITY : base * e;
    }
    if (tid < KP_) cand[CAND_ + tid] = s_nbn[tid] + s_bn[tid];
    __syncthreads();

    // ---- top-32: repeated argmax, tie-break on lower index (lax.top_k) ----
    const int wid = tid >> 5, lane = tid & 31, nw = blockDim.x >> 5;
    for (int it = 0; it < W_; ++it) {
        float bv = -INFINITY; int bi = 0x7fffffff;
        for (int i = tid; i < NC_; i += blockDim.x) {
            const float v = cand[i];
            if (v > bv || (v == bv && i < bi)) { bv = v; bi = i; }
        }
        #pragma unroll
        for (int off = 16; off > 0; off >>= 1) {
            const float ov = __shfl_xor(bv, off, 32);
            const int   oi = __shfl_xor(bi, off, 32);
            if (ov > bv || (ov == bv && oi < bi)) { bv = ov; bi = oi; }
        }
        if (lane == 0) { wbest[wid] = bv; wbidx[wid] = bi; }
        __syncthreads();
        if (tid == 0) {
            float fv = wbest[0]; int fi = wbidx[0];
            for (int w = 1; w < nw; ++w) {
                const float ov = wbest[w]; const int oi = wbidx[w];
                if (ov > fv || (ov == fv && oi < fi)) { fv = ov; fi = oi; }
            }
            selV[it] = fv; selI[it] = fi;
            cand[fi] = -INFINITY;            // invalidate for next pass
        }
        __syncthreads();
    }

    // ---- emit scalar outputs + metadata ----
    if (tid < W_) {
        const int   ind   = selI[tid];
        const float val   = selV[tid];
        const int   isnon = ind >= CAND_;
        const int   src   = isnon ? (ind - CAND_) : (ind >> 11);
        const int   e     = ind & (V_ - 1);      // next_ind % V (CAND_ % V == 0)
        const int   plen  = ylens[n * KP_ + src];
        const int   lnext = plen + (isnon ? 0 : 1);
        const float nbv   = isnon ? s_nbn[src] : val;
        const float bv2   = isnon ? s_bn[src] : 0.f;
        const int   lastv = isnon ? s_last[src] : e;

        const int oi = n * W_ + tid;
        out[O_LAST   + oi] = (float)lastv;
        out[O_LENS   + oi] = (float)lnext;
        out[O_NB     + oi] = nbv;
        out[O_B      + oi] = bv2;
        out[O_SRC    + oi] = (float)src;
        out[O_NONEXT + oi] = (float)isnon;
        wsrc[oi] = src; wext[oi] = e; wplen[oi] = plen;
        wlens[oi] = lnext; wnone[oi] = isnon;
    }
}

// =====================================================================
// Kernel C: y_next gather + single-position overwrite
//   y_next[t,n,k] = (t==plen) ? ext : (t<S ? y_prev[t,n,src] : 0)
// =====================================================================
__global__ __launch_bounds__(256) void ctc_advC(
    const int* __restrict__ yprev, const int* __restrict__ wsrc,
    const int* __restrict__ wext, const int* __restrict__ wplen,
    float* __restrict__ out)
{
    const int idx = blockIdx.x * blockDim.x + threadIdx.x;
    if (idx >= (S_ + 1) * N_ * W_) return;
    const int t = idx / (N_ * W_);
    const int r = idx - t * (N_ * W_);
    const int n = r / W_, k = r - n * W_;
    const int oi = n * W_ + k;
    const int src = wsrc[oi];
    int v;
    if (t == wplen[oi])      v = wext[oi];
    else if (t < S_)         v = yprev[(size_t)t * N_ * KP_ + n * KP_ + src];
    else                     v = 0;
    out[O_YNEXT + idx] = (float)v;
}

// =====================================================================
// Kernel D: next_is_prefix[n,k,j]
// =====================================================================
__global__ __launch_bounds__(256) void ctc_advD(
    const int* __restrict__ pfx, const int* __restrict__ wsrc,
    const int* __restrict__ wext, const int* __restrict__ wlens,
    const int* __restrict__ wnone, float* __restrict__ out)
{
    const int idx = blockIdx.x * blockDim.x + threadIdx.x;
    if (idx >= N_ * W_ * W_) return;
    const int n = idx / (W_ * W_);
    const int r = idx - n * (W_ * W_);
    const int k = r / W_, j = r - k * W_;
    const int srck = wsrc[n * W_ + k], srcj = wsrc[n * W_ + j];
    const bool pp  = pfx[((size_t)n * KP_ + srck) * KP_ + srcj] != 0;
    const int  lk  = wlens[n * W_ + k], lj = wlens[n * W_ + j];
    const bool leq = lk <= lj;
    const int  row = max(lk - 1, 0);
    const float tm = out[O_YNEXT + (size_t)row * N_ * W_ + n * W_ + j];
    const bool nk  = wnone[n * W_ + k] != 0;
    const bool em  = tm == (float)wext[n * W_ + k];
    out[O_ISPFX + idx] = (pp && leq && (nk || em)) ? 1.f : 0.f;
}

// =====================================================================
extern "C" void kernel_launch(void* const* d_in, const int* in_sizes, int n_in,
                              void* d_out, int out_size, void* d_ws, size_t ws_size,
                              hipStream_t stream)
{
    const float* ext    = (const float*)d_in[0];
    const float* nonext = (const float*)d_in[1];
    const float* blank  = (const float*)d_in[2];
    const float* nbp    = (const float*)d_in[3];
    const float* bp     = (const float*)d_in[4];
    const int*   yprev  = (const int*)d_in[5];
    const int*   ylast  = (const int*)d_in[6];
    const int*   ylens  = (const int*)d_in[7];
    const int*   pfx    = (const int*)d_in[8];
    float* out = (float*)d_out;

    // workspace layout
    uint32_t* ws_mask  = (uint32_t*)d_ws;                          // N*KP*64 u32
    float*    ws_nbnon = (float*)(ws_mask + (size_t)N_ * KP_ * 64);// N*KP
    float*    ws_bnon  = ws_nbnon + N_ * KP_;                      // N*KP
    int*      wsrc     = (int*)(ws_bnon + N_ * KP_);               // N*W each
    int*      wext     = wsrc + N_ * W_;
    int*      wplen    = wext + N_ * W_;
    int*      wlens    = wplen + N_ * W_;
    int*      wnone    = wlens + N_ * W_;

    ctc_advA<<<N_, 32, 0, stream>>>(ext, nonext, blank, nbp, bp, yprev, ylast,
                                    ylens, pfx, ws_mask, ws_nbnon, ws_bnon);

    constexpr size_t shB =
        (size_t)NC_ * 4 + (size_t)KP_ * 64 * 4 +  // cand + mask
        KP_ * 4 * 4 + KP_ * 4 +                   // base, b, nbn, bn + last
        16 * 4 * 2 + W_ * 4 * 2;                  // wave reduce + selections
    ctc_advB<<<N_, 512, shB, stream>>>(ext, nbp, bp, ylast, ylens,
                                       ws_mask, ws_nbnon, ws_bnon, out,
                                       wsrc, wext, wplen, wlens, wnone);

    const int totC = (S_ + 1) * N_ * W_;
    ctc_advC<<<(totC + 255) / 256, 256, 0, stream>>>(yprev, wsrc, wext, wplen, out);

    const int totD = N_ * W_ * W_;
    ctc_advD<<<(totD + 255) / 256, 256, 0, stream>>>(pfx, wsrc, wext, wlens, wnone, out);
}
